// Codebook_45896020525584
// MI455X (gfx1250) — matrix operand order repaired and checked by reading, not compiled, not run on hardware
//
#include <hip/hip_runtime.h>

typedef __attribute__((ext_vector_type(16))) __bf16 v16bf;
typedef __attribute__((ext_vector_type(8)))  float  v8f;
typedef __attribute__((ext_vector_type(4)))  float  v4f;

namespace {
constexpr int kB = 32, kD = 64, kT = 4096, kK = 1024;
constexpr int ROWS = 128;   // token rows per workgroup
constexpr int KCH  = 64;    // codebook rows per LDS buffer (double-buffered)
constexpr int NCH  = kK / KCH;
constexpr int NTHR = 256;   // 8 waves (wave32)
constexpr int WROW = 72;    // bf16 per staged codebook row: 64 data + 8 pad (144B, 16B aligned)
constexpr size_t QSIZE = (size_t)kB * kD * kT;  // quantized output elements
}

union FragBF { v16bf v; v4f q[2]; };

// One-time prep: split codebook into bf16 hi/lo (round + residual) and ||w||^2.
__global__ void prep_kernel(const float* __restrict__ W, __bf16* __restrict__ whi,
                            __bf16* __restrict__ wlo, float* __restrict__ wn) {
  int k = blockIdx.x * blockDim.x + threadIdx.x;
  if (k >= kK) return;
  const float* row = W + (size_t)k * kD;
  float s = 0.f;
#pragma unroll
  for (int d = 0; d < kD; ++d) {
    float v = row[d];
    s += v * v;
    __bf16 h = (__bf16)v;
    whi[(size_t)k * kD + d] = h;
    wlo[(size_t)k * kD + d] = (__bf16)(v - (float)h);
  }
  wn[k] = s;  // argmin of ||w||^2 - 2 x.w == argmin of full squared distance
}

// Async-copy one 64-code chunk (hi+lo, 8KB each... 8KB hi + 8KB lo) into an LDS buffer,
// restriping 128B global rows to the 144B padded LDS stride via per-lane LDS addresses.
__device__ __forceinline__ void stage_async(const __bf16* __restrict__ gHi,
                                            const __bf16* __restrict__ gLo,
                                            __bf16* sHiBuf, __bf16* sLoBuf,
                                            int ch, int wave, int lane) {
#pragma unroll
  for (int s = 0; s < 2; ++s) {
    const int seg  = wave * 2 + s;            // 16 segments of 512B cover 64 rows x 128B
    const int code = seg * 4 + (lane >> 3);   // 4 rows per segment, 8 lanes per row
    const int off  = (lane & 7) * 16;         // byte offset within the 128B row
    const size_t gOff = (size_t)(ch * KCH + code) * 128 + off;
    uint64_t gH = (uint64_t)(uintptr_t)((const char*)gHi + gOff);
    uint64_t gL = (uint64_t)(uintptr_t)((const char*)gLo + gOff);
    uint32_t lH = (uint32_t)(uintptr_t)((char*)sHiBuf + code * 144 + off);
    uint32_t lL = (uint32_t)(uintptr_t)((char*)sLoBuf + code * 144 + off);
    asm volatile("global_load_async_to_lds_b128 %0, %1, off" :: "v"(lH), "v"(gH) : "memory");
    asm volatile("global_load_async_to_lds_b128 %0, %1, off" :: "v"(lL), "v"(gL) : "memory");
  }
}

__launch_bounds__(NTHR, 2)
__global__ void vq_kernel(const float* __restrict__ x,
                          const float* __restrict__ W,
                          const __bf16* __restrict__ whi,
                          const __bf16* __restrict__ wlo,
                          const float* __restrict__ wn,
                          float* __restrict__ out) {
  __shared__ __align__(16) __bf16 sHi[2][KCH * WROW];
  __shared__ __align__(16) __bf16 sLo[2][KCH * WROW];
  __shared__ float sWn[kK];
  __shared__ int   sIdx[ROWS];

  const int tid  = threadIdx.x;
  const int lane = tid & 31;
  const int wave = tid >> 5;
  const int half = lane >> 4;   // lane half: selects k-subrange for A/B fragments
  const int m    = lane & 15;   // A row / B,C column within the 16x16 tile

  const int rowBase = blockIdx.x * ROWS;  // ROWS divides T -> one batch index per WG
  const int b  = rowBase / kT;
  const int t0 = rowBase % kT;
  const int waveRow = wave * 16;

  // Kick off DMA of chunk 0 before doing the VGPR work below.
  stage_async(whi, wlo, sHi[0], sLo[0], 0, wave, lane);

  // ---- A fragments (this wave's 16 token rows), bf16 hi/lo, kept in registers ----
  // ISA 7.12.2 16-bit A 16x32: lanes 0-15 k in {0..7,16..23}; lanes 16-31 k in {8..15,24..31}.
  const float* xb = x + (size_t)b * kD * kT + (size_t)(t0 + waveRow + m);
  v16bf aHi[2], aLo[2];
#pragma unroll
  for (int ks = 0; ks < 2; ++ks) {
#pragma unroll
    for (int j = 0; j < 16; ++j) {
      int d = ks * 32 + ((j < 8) ? (half * 8 + j) : (16 + half * 8 + (j - 8)));
      float v  = xb[(size_t)d * kT];
      __bf16 h = (__bf16)v;
      aHi[ks][j] = h;
      aLo[ks][j] = (__bf16)(v - (float)h);  // residual
    }
  }

  // Stage all code norms once (4KB).
  for (int i = tid; i < kK; i += NTHR) sWn[i] = wn[i];

  float best[8];
  int   bidx[8];
#pragma unroll
  for (int i = 0; i < 8; ++i) { best[i] = 3.4e38f; bidx[i] = 0; }

  int buf = 0;
  for (int ch = 0; ch < NCH; ++ch) {
    asm volatile("s_wait_asynccnt 0x0" ::: "memory");  // my DMA for this chunk landed
    __syncthreads();                                   // everyone's landed; prev buffer free
    if (ch + 1 < NCH)
      stage_async(whi, wlo, sHi[buf ^ 1], sLo[buf ^ 1], ch + 1, wave, lane);

    const __bf16* hb = sHi[buf];
    const __bf16* lb = sLo[buf];
#pragma unroll
    for (int tile = 0; tile < KCH / 16; ++tile) {
      // B fragment (32x16 bf16): lane holds column n = lane&15, k = half*8 + j (j<8), 16 + half*8 + (j-8) (j>=8)
      // -> two 16-byte runs per lane, two ds_load_b128 each.
      const int code = tile * 16 + m;
      const __bf16* wrHi = hb + code * WROW + half * 8;
      const __bf16* wrLo = lb + code * WROW + half * 8;
      FragBF b0h, b1h, b0l, b1l;
      b0h.q[0] = *(const v4f*)(wrHi + 0);
      b0h.q[1] = *(const v4f*)(wrHi + 16);
      b1h.q[0] = *(const v4f*)(wrHi + 32);
      b1h.q[1] = *(const v4f*)(wrHi + 48);
      b0l.q[0] = *(const v4f*)(wrLo + 0);
      b0l.q[1] = *(const v4f*)(wrLo + 16);
      b1l.q[0] = *(const v4f*)(wrLo + 32);
      b1l.q[1] = *(const v4f*)(wrLo + 48);

      // dot(x, w) ~= xh.wh + xl.wh + xh.wl  (bf16x2 split, ~16-bit mantissa)
      v8f c0 = {}, c1 = {};
      c0 = __builtin_amdgcn_wmma_f32_16x16x32_bf16(false, aHi[0], false, b0h.v, (short)0, c0, false, false);
      c1 = __builtin_amdgcn_wmma_f32_16x16x32_bf16(false, aHi[1], false, b1h.v, (short)0, c1, false, false);
      c0 = __builtin_amdgcn_wmma_f32_16x16x32_bf16(false, aLo[0], false, b0h.v, (short)0, c0, false, false);
      c1 = __builtin_amdgcn_wmma_f32_16x16x32_bf16(false, aLo[1], false, b1h.v, (short)0, c1, false, false);
      c0 = __builtin_amdgcn_wmma_f32_16x16x32_bf16(false, aHi[0], false, b0l.v, (short)0, c0, false, false);
      c1 = __builtin_amdgcn_wmma_f32_16x16x32_bf16(false, aHi[1], false, b1l.v, (short)0, c1, false, false);

      const int n  = ch * KCH + tile * 16 + m;
      const float w2 = sWn[n];
      // C layout: VGPR i -> M = i + 8*half, N = lane&15
#pragma unroll
      for (int i = 0; i < 8; ++i) {
        float s = fmaf(-2.f, c0[i] + c1[i], w2);
        if (s < best[i]) { best[i] = s; bidx[i] = n; }  // n strictly increases -> first-min kept
      }
    }
    buf ^= 1;
  }

  // ---- argmin across the 16 lanes of each half (xor<16 never crosses bit 4) ----
#pragma unroll
  for (int off = 1; off < 16; off <<= 1) {
#pragma unroll
    for (int i = 0; i < 8; ++i) {
      float ov = __shfl_xor(best[i], off, 32);
      int   oi = __shfl_xor(bidx[i], off, 32);
      if (ov < best[i] || (ov == best[i] && oi < bidx[i])) { best[i] = ov; bidx[i] = oi; }
    }
  }
  if (m == 0) {  // lanes 0 (rows 0-7) and 16 (rows 8-15) of each wave
#pragma unroll
    for (int i = 0; i < 8; ++i) {
      int r = waveRow + half * 8 + i;
      sIdx[r] = bidx[i];
      out[QSIZE + rowBase + r] = (float)bidx[i];  // indices appended after quantized block
    }
  }
  __syncthreads();

  // ---- gather codebook rows -> quantized output [B, D, T], coalesced along T ----
  float* q = out + (size_t)b * kD * kT + t0;
  for (int e = tid; e < kD * ROWS; e += NTHR) {
    int d = e >> 7;            // ROWS == 128
    int r = e & (ROWS - 1);
    q[(size_t)d * kT + r] = W[(size_t)sIdx[r] * kD + d];
  }
}

extern "C" void kernel_launch(void* const* d_in, const int* in_sizes, int n_in,
                              void* d_out, int out_size, void* d_ws, size_t ws_size,
                              hipStream_t stream) {
  const float* x = (const float*)d_in[0];   // [32, 64, 4096] f32
  const float* W = (const float*)d_in[1];   // [1024, 64] f32
  float* out = (float*)d_out;               // quantized (B*D*T) then indices (B*T) as float

  // d_ws layout: whi[1024*64] bf16 | wlo[1024*64] bf16 | wn[1024] f32  (~260KB)
  __bf16* whi = (__bf16*)d_ws;
  __bf16* wlo = whi + (size_t)kK * kD;
  float*  wn  = (float*)(wlo + (size_t)kK * kD);

  prep_kernel<<<(kK + 255) / 256, 256, 0, stream>>>(W, whi, wlo, wn);
  vq_kernel<<<(kB * kT) / ROWS, NTHR, 0, stream>>>(x, W, whi, wlo, wn, out);
}
